// DecoderBlock_44650480009686
// MI455X (gfx1250) — matrix operand order repaired and checked by reading, not compiled, not run on hardware
//
#include <hip/hip_runtime.h>
#include <hip/hip_bf16.h>
#include <math.h>

typedef __attribute__((ext_vector_type(16))) _Float16 v16h;
typedef __attribute__((ext_vector_type(8)))  float    v8f;
typedef __attribute__((ext_vector_type(4)))  int      v4i;

#define D_MODEL 768
#define SEQ 4096
#define HEADS 12
#define HD 64
#define HIDDEN 3072
#define QKV_N (3 * D_MODEL)   // 2304

// fragment = 16 halves per lane = two 16B chunks -> b128 LDS/global loads
struct FragH { union { v16h v; uint4 q[2]; }; };

__device__ __forceinline__ v8f wmma16(v16h a, v16h b, v8f c) {
    return __builtin_amdgcn_wmma_f32_16x16x32_f16(false, a, false, b,
                                                  (short)0, c, false, false);
}

// ---- CDNA5 async global->LDS (ASYNCcnt) with safe fallback --------------
#if defined(__has_builtin)
#if __has_builtin(__builtin_amdgcn_global_load_async_to_lds_b128)
#define HAVE_ASYNC_LDS 1
#endif
#endif
#ifndef HAVE_ASYNC_LDS
#define HAVE_ASYNC_LDS 0
#endif

typedef __attribute__((address_space(1))) v4i* gptr_v4i;
typedef __attribute__((address_space(3))) v4i* lptr_v4i;

__device__ __forceinline__ void async_ld16(void* lds_dst, const void* gsrc) {
#if HAVE_ASYNC_LDS
    __builtin_amdgcn_global_load_async_to_lds_b128(
        (gptr_v4i)gsrc, (lptr_v4i)lds_dst, 0, 0);
#else
    *(uint4*)lds_dst = *(const uint4*)gsrc;
#endif
}
__device__ __forceinline__ void async_wait_all() {
#if HAVE_ASYNC_LDS
    asm volatile("s_wait_asynccnt 0x0" ::: "memory");
#endif
}

// ---------------------------------------------------------------------------
// f32 -> f16 conversion (weights)
// ---------------------------------------------------------------------------
__global__ void cvt_f32_to_f16(const float* __restrict__ in,
                               _Float16* __restrict__ out, int n) {
    int i = blockIdx.x * 256 + threadIdx.x;
    if (i < n) out[i] = (_Float16)in[i];
}

// ---------------------------------------------------------------------------
// LayerNorm over 768 -> f16.  One row per block; wave32 shfl reductions,
// only two barriers per row.
// ---------------------------------------------------------------------------
__global__ void ln_kernel(const float* __restrict__ x,
                          const float* __restrict__ gamma,
                          const float* __restrict__ beta,
                          _Float16* __restrict__ out) {
    __shared__ float psum[8];
    __shared__ float pvar[8];
    const int row = blockIdx.x;
    const int t = threadIdx.x;
    const int lane = t & 31, wv = t >> 5;
    const float* xr = x + (size_t)row * D_MODEL;
    float v0 = xr[t], v1 = xr[t + 256], v2 = xr[t + 512];
    float s = v0 + v1 + v2;
    s += __shfl_xor(s, 1);  s += __shfl_xor(s, 2);
    s += __shfl_xor(s, 4);  s += __shfl_xor(s, 8);
    s += __shfl_xor(s, 16);
    if (lane == 0) psum[wv] = s;
    __syncthreads();
    float mean = (psum[0] + psum[1] + psum[2] + psum[3] +
                  psum[4] + psum[5] + psum[6] + psum[7]) * (1.0f / 768.0f);
    float d0 = v0 - mean, d1 = v1 - mean, d2 = v2 - mean;
    float q = d0 * d0 + d1 * d1 + d2 * d2;
    q += __shfl_xor(q, 1);  q += __shfl_xor(q, 2);
    q += __shfl_xor(q, 4);  q += __shfl_xor(q, 8);
    q += __shfl_xor(q, 16);
    if (lane == 0) pvar[wv] = q;
    __syncthreads();
    float rstd = rsqrtf((pvar[0] + pvar[1] + pvar[2] + pvar[3] +
                         pvar[4] + pvar[5] + pvar[6] + pvar[7]) *
                            (1.0f / 768.0f) + 1e-5f);
    _Float16* orow = out + (size_t)row * D_MODEL;
    orow[t]       = (_Float16)(gamma[t]       * (d0 * rstd) + beta[t]);
    orow[t + 256] = (_Float16)(gamma[t + 256] * (d1 * rstd) + beta[t + 256]);
    orow[t + 512] = (_Float16)(gamma[t + 512] * (d2 * rstd) + beta[t + 512]);
}

// ---------------------------------------------------------------------------
// f16 WMMA GEMM: C[M,N] = A[M,K] @ B[K,N] + bias (+residual)(+GELU)
// 128x128 block tile, BK=32, 8 waves each computing 64x32 (8 accumulators).
// Double-buffered LDS; A staged via async global->LDS (ASYNCcnt), B staged
// transposed through VGPRs; ONE barrier per K-step.
// ---------------------------------------------------------------------------
template <bool GELU, bool RES, bool W32, bool W16>
__global__ __launch_bounds__(256)
void gemm_f16_wmma(const _Float16* __restrict__ A,
                   const _Float16* __restrict__ B,
                   const float* __restrict__ bias,
                   const float* __restrict__ residual,
                   float* __restrict__ out32,
                   _Float16* __restrict__ out16,
                   int M, int N, int K) {
    constexpr int BM = 128, BN = 128, BK = 32;
    __shared__ alignas(16) _Float16 As[2][BM][BK];       // 2 x 8 KB
    __shared__ alignas(16) _Float16 Bs[2][BN][BK + 8];   // 2 x 10 KB

    const int t = threadIdx.x;
    const int wv = t >> 5;
    const int lane = t & 31;
    const int lo = lane & 15;
    const int hi = lane >> 4;
    const int mw = (wv >> 2) * 64;
    const int nw = (wv & 3) * 32;
    const int m0 = blockIdx.y * BM;
    const int n0 = blockIdx.x * BN;

    const int ar = t >> 1, ac = (t & 1) * 16;   // A staging: 2 x b128
    const int br = t >> 3, bc = (t & 7) * 16;   // B staging: 16 halves

    v8f c[4][2];
#pragma unroll
    for (int mt = 0; mt < 4; ++mt)
#pragma unroll
        for (int nt = 0; nt < 2; ++nt) c[mt][nt] = (v8f){};

    {   // prologue: stage tile 0 into buffer 0
        const _Float16* ap = A + (size_t)(m0 + ar) * K + ac;
        async_ld16(&As[0][ar][ac], ap);
        async_ld16(&As[0][ar][ac + 8], ap + 8);
        const _Float16* bp = B + (size_t)br * N + n0 + bc;
        uint4 b0g = *(const uint4*)bp, b1g = *(const uint4*)(bp + 8);
        const _Float16* h0 = (const _Float16*)&b0g;
        const _Float16* h1 = (const _Float16*)&b1g;
#pragma unroll
        for (int j = 0; j < 8; ++j) {
            Bs[0][bc + j][br]     = h0[j];
            Bs[0][bc + 8 + j][br] = h1[j];
        }
    }
    async_wait_all();
    __syncthreads();

    const int KT = K / BK;
    for (int kt = 0; kt < KT; ++kt) {
        const int buf = kt & 1;
        const bool more = (kt + 1 < KT);
        uint4 b0g, b1g;
        if (more) {   // prefetch tile kt+1: A async into back buffer, B->regs
            const int k0 = (kt + 1) * BK;
            const _Float16* ap = A + (size_t)(m0 + ar) * K + k0 + ac;
            async_ld16(&As[buf ^ 1][ar][ac], ap);
            async_ld16(&As[buf ^ 1][ar][ac + 8], ap + 8);
            const _Float16* bp = B + (size_t)(k0 + br) * N + n0 + bc;
            b0g = *(const uint4*)bp;  b1g = *(const uint4*)(bp + 8);
        }

        FragH a[4], b[2];
#pragma unroll
        for (int mt = 0; mt < 4; ++mt) {
            const _Float16* p = &As[buf][mw + mt * 16 + lo][hi * 8];
            a[mt].q[0] = *(const uint4*)p;
            a[mt].q[1] = *(const uint4*)(p + 16);
        }
#pragma unroll
        for (int nt = 0; nt < 2; ++nt) {
            const _Float16* p = &Bs[buf][nw + nt * 16 + lo][hi * 8];
            b[nt].q[0] = *(const uint4*)p;
            b[nt].q[1] = *(const uint4*)(p + 16);
        }
#pragma unroll
        for (int mt = 0; mt < 4; ++mt)
#pragma unroll
            for (int nt = 0; nt < 2; ++nt)
                c[mt][nt] = wmma16(a[mt].v, b[nt].v, c[mt][nt]);

        if (more) {   // transpose-store B tile kt+1 into back buffer
            const _Float16* h0 = (const _Float16*)&b0g;
            const _Float16* h1 = (const _Float16*)&b1g;
#pragma unroll
            for (int j = 0; j < 8; ++j) {
                Bs[buf ^ 1][bc + j][br]     = h0[j];
                Bs[buf ^ 1][bc + 8 + j][br] = h1[j];
            }
        }
        async_wait_all();
        __syncthreads();
    }

    // epilogue (compile-time specialized)
#pragma unroll
    for (int mt = 0; mt < 4; ++mt)
#pragma unroll
        for (int nt = 0; nt < 2; ++nt) {
            const int col = n0 + nw + nt * 16 + lo;
            const float bv = bias[col];
#pragma unroll
            for (int i = 0; i < 8; ++i) {
                const int row = m0 + mw + mt * 16 + i + 8 * hi;
                float v = c[mt][nt][i] + bv;
                if constexpr (RES)  v += residual[(size_t)row * N + col];
                if constexpr (GELU) v = 0.5f * v * (1.0f + erff(v * 0.70710678118f));
                if constexpr (W32)  out32[(size_t)row * N + col] = v;
                if constexpr (W16)  out16[(size_t)row * N + col] = (_Float16)v;
            }
        }
}

// ---------------------------------------------------------------------------
// Flash-style causal attention, 64-key tiles.  qkv f16: [S, H*192], per head
// q/k/v at +0/+64/+128.  4 waves x 16 query rows; K staged via async
// global->LDS, V transposed ([d][key]); online softmax with a scalar-branch
// fast path for fully-unmasked tiles; 16 WMMAs per tile.
// ---------------------------------------------------------------------------
__global__ __launch_bounds__(128)
void attn_flash_wmma(const _Float16* __restrict__ qkv,
                     _Float16* __restrict__ out16) {
    __shared__ alignas(16) _Float16 Ks[64][64];        // [key][d]   8 KB
    __shared__ alignas(16) _Float16 Vst[64][64 + 8];   // [d][key]   9 KB
    __shared__ alignas(16) _Float16 Ps[4][16][64];     // per-wave P 8 KB

    const int t = threadIdx.x;
    const int w = t >> 5;
    const int lane = t & 31;
    const int lo = lane & 15;
    const int hi = lane >> 4;
    const int head = blockIdx.y;
    const int qb = blockIdx.x * 64;
    const int q0 = qb + w * 16;

    FragH qa0, qa1;   // Q fragments, d-chunks [0,32) and [32,64)
    {
        const _Float16* qp =
            qkv + (size_t)(q0 + lo) * QKV_N + head * 192 + hi * 8;
        qa0.q[0] = *(const uint4*)qp;
        qa0.q[1] = *(const uint4*)(qp + 16);
        qa1.q[0] = *(const uint4*)(qp + 32);
        qa1.q[1] = *(const uint4*)(qp + 48);
    }

    v8f o0 = {}, o1 = {}, o2 = {}, o3 = {};
    float mrow[8], lrow[8];
#pragma unroll
    for (int i = 0; i < 8; ++i) { mrow[i] = -1e30f; lrow[i] = 0.0f; }

    const int ntiles = qb / 64 + 1;
    for (int kt = 0; kt < ntiles; ++kt) {
        const int k0 = kt * 64;
        // stage K (async, row-major) and V (transposed through VGPRs)
        for (int cch = t; cch < 512; cch += 128) {
            int r = cch >> 3, col = (cch & 7) * 8;
            size_t base = (size_t)(k0 + r) * QKV_N + head * 192;
            async_ld16(&Ks[r][col], qkv + base + 64 + col);
            uint4 vv = *(const uint4*)(qkv + base + 128 + col);
            const _Float16* vh = (const _Float16*)&vv;
#pragma unroll
            for (int j = 0; j < 8; ++j) Vst[col + j][r] = vh[j];
        }
        async_wait_all();
        __syncthreads();

        // S = Q @ K^T : 4 key subtiles x 2 d-chunks = 8 WMMAs
        v8f s[4];
#pragma unroll
        for (int j = 0; j < 4; ++j) {
            v8f acc = {};
#pragma unroll
            for (int ch = 0; ch < 2; ++ch) {
                const _Float16* p = &Ks[j * 16 + lo][ch * 32 + hi * 8];
                FragH f;
                f.q[0] = *(const uint4*)p;
                f.q[1] = *(const uint4*)(p + 16);
                acc = wmma16(ch ? qa1.v : qa0.v, f.v, acc);
            }
            s[j] = acc;
        }

        // wave-uniform mask predicate -> scalar branch skips compare chains
        int nm = ((k0 + 63) > q0) ? 1 : 0;
        nm = __builtin_amdgcn_readfirstlane(nm);

#pragma unroll
        for (int i = 0; i < 8; ++i) {
            const int row = q0 + i + 8 * hi;
            float a[4];
#pragma unroll
            for (int j = 0; j < 4; ++j) a[j] = s[j][i] * 0.125f;
            if (nm) {
#pragma unroll
                for (int j = 0; j < 4; ++j)
                    if (k0 + j * 16 + lo > row) a[j] = -1e9f;
            }
            float mx = fmaxf(fmaxf(a[0], a[1]), fmaxf(a[2], a[3]));
            mx = fmaxf(mx, __shfl_xor(mx, 1));
            mx = fmaxf(mx, __shfl_xor(mx, 2));
            mx = fmaxf(mx, __shfl_xor(mx, 4));
            mx = fmaxf(mx, __shfl_xor(mx, 8));
            const float mnew = fmaxf(mrow[i], mx);
            const float fac = __expf(mrow[i] - mnew);
            float rs = 0.0f;
#pragma unroll
            for (int j = 0; j < 4; ++j) { a[j] = __expf(a[j] - mnew); rs += a[j]; }
            rs += __shfl_xor(rs, 1);
            rs += __shfl_xor(rs, 2);
            rs += __shfl_xor(rs, 4);
            rs += __shfl_xor(rs, 8);
            lrow[i] = lrow[i] * fac + rs;
            mrow[i] = mnew;
            o0[i] *= fac; o1[i] *= fac; o2[i] *= fac; o3[i] *= fac;
#pragma unroll
            for (int j = 0; j < 4; ++j)
                Ps[w][i + 8 * hi][j * 16 + lo] = (_Float16)a[j];
        }
        asm volatile("s_wait_dscnt 0" ::: "memory");   // intra-wave LDS RAW

        FragH pa0, pa1;   // P in A-layout, key-chunks [0,32) and [32,64)
        {
            const _Float16* pp = &Ps[w][lo][hi * 8];
            pa0.q[0] = *(const uint4*)pp;
            pa0.q[1] = *(const uint4*)(pp + 16);
            pa1.q[0] = *(const uint4*)(pp + 32);
            pa1.q[1] = *(const uint4*)(pp + 48);
        }
        // O += P @ V : 4 d-tiles x 2 key-chunks = 8 WMMAs
#pragma unroll
        for (int f = 0; f < 4; ++f) {
            const _Float16* vp = &Vst[f * 16 + lo][hi * 8];
            FragH vb0, vb1;
            vb0.q[0] = *(const uint4*)vp;
            vb0.q[1] = *(const uint4*)(vp + 16);
            vb1.q[0] = *(const uint4*)(vp + 32);
            vb1.q[1] = *(const uint4*)(vp + 48);
            v8f oo = (f == 0) ? o0 : (f == 1) ? o1 : (f == 2) ? o2 : o3;
            oo = wmma16(pa0.v, vb0.v, oo);
            oo = wmma16(pa1.v, vb1.v, oo);
            if (f == 0) o0 = oo; else if (f == 1) o1 = oo;
            else if (f == 2) o2 = oo; else o3 = oo;
        }
        __syncthreads();
    }

#pragma unroll
    for (int f = 0; f < 4; ++f) {
        v8f oo = (f == 0) ? o0 : (f == 1) ? o1 : (f == 2) ? o2 : o3;
#pragma unroll
        for (int i = 0; i < 8; ++i) {
            const int row = q0 + i + 8 * hi;
            const int d = f * 16 + lo;
            out16[(size_t)row * D_MODEL + head * HD + d] =
                (_Float16)(oo[i] / lrow[i]);
        }
    }
}

// ---------------------------------------------------------------------------
// Orchestration
// ---------------------------------------------------------------------------
extern "C" void kernel_launch(void* const* d_in, const int* in_sizes, int n_in,
                              void* d_out, int out_size, void* d_ws, size_t ws_size,
                              hipStream_t stream) {
    (void)in_sizes; (void)n_in; (void)out_size; (void)ws_size;
    const float* x      = (const float*)d_in[0];
    const float* gamma1 = (const float*)d_in[1];
    const float* beta1  = (const float*)d_in[2];
    const float* Wqkv   = (const float*)d_in[3];
    const float* bqkv   = (const float*)d_in[4];
    const float* Wo     = (const float*)d_in[5];
    const float* bo     = (const float*)d_in[6];
    const float* gamma2 = (const float*)d_in[7];
    const float* beta2  = (const float*)d_in[8];
    const float* W1     = (const float*)d_in[9];
    const float* b1     = (const float*)d_in[10];
    const float* W2     = (const float*)d_in[11];
    const float* b2     = (const float*)d_in[12];
    float* out = (float*)d_out;

    char* ws = (char*)d_ws;
    size_t off = 0;
    auto carve = [&](size_t bytes) -> char* {
        char* p = ws + off;
        off = (off + bytes + 255) & ~(size_t)255;
        return p;
    };
    _Float16* wqkv16 = (_Float16*)carve((size_t)D_MODEL * QKV_N * 2);
    _Float16* wo16   = (_Float16*)carve((size_t)D_MODEL * D_MODEL * 2);
    _Float16* w1h    = (_Float16*)carve((size_t)D_MODEL * HIDDEN * 2);
    _Float16* w2h    = (_Float16*)carve((size_t)HIDDEN * D_MODEL * 2);
    _Float16* xn16   = (_Float16*)carve((size_t)SEQ * D_MODEL * 2);
    _Float16* qkv16  = (_Float16*)carve((size_t)SEQ * QKV_N * 2);
    _Float16* attn16 = (_Float16*)carve((size_t)SEQ * D_MODEL * 2);
    float*    y1     = (float*)   carve((size_t)SEQ * D_MODEL * 4);
    _Float16* h16    = (_Float16*)carve((size_t)SEQ * HIDDEN * 2);

    {   // weight conversion f32 -> f16
        int n;
        n = D_MODEL * QKV_N;
        cvt_f32_to_f16<<<(n + 255) / 256, 256, 0, stream>>>(Wqkv, wqkv16, n);
        n = D_MODEL * D_MODEL;
        cvt_f32_to_f16<<<(n + 255) / 256, 256, 0, stream>>>(Wo, wo16, n);
        n = D_MODEL * HIDDEN;
        cvt_f32_to_f16<<<(n + 255) / 256, 256, 0, stream>>>(W1, w1h, n);
        n = HIDDEN * D_MODEL;
        cvt_f32_to_f16<<<(n + 255) / 256, 256, 0, stream>>>(W2, w2h, n);
    }

    // LN1
    ln_kernel<<<SEQ, 256, 0, stream>>>(x, gamma1, beta1, xn16);

    // QKV projection -> f16
    gemm_f16_wmma<false, false, false, true>
        <<<dim3(QKV_N / 128, SEQ / 128), 256, 0, stream>>>(
            xn16, wqkv16, bqkv, nullptr, nullptr, qkv16,
            SEQ, QKV_N, D_MODEL);

    // causal flash attention
    attn_flash_wmma<<<dim3(SEQ / 64, HEADS), 128, 0, stream>>>(qkv16, attn16);

    // O-proj + residual -> y1 (f32)
    gemm_f16_wmma<false, true, true, false>
        <<<dim3(D_MODEL / 128, SEQ / 128), 256, 0, stream>>>(
            attn16, wo16, bo, x, y1, nullptr,
            SEQ, D_MODEL, D_MODEL);

    // LN2 (reuse xn16)
    ln_kernel<<<SEQ, 256, 0, stream>>>(y1, gamma2, beta2, xn16);

    // FFN up + exact GELU -> f16
    gemm_f16_wmma<true, false, false, true>
        <<<dim3(HIDDEN / 128, SEQ / 128), 256, 0, stream>>>(
            xn16, w1h, b1, nullptr, nullptr, h16,
            SEQ, HIDDEN, D_MODEL);

    // FFN down + residual -> d_out (f32)
    gemm_f16_wmma<false, true, true, false>
        <<<dim3(D_MODEL / 128, SEQ / 128), 256, 0, stream>>>(
            h16, w2h, b2, y1, out, nullptr,
            SEQ, D_MODEL, HIDDEN);
}
